// LayerEdgeGCNConv_7430293422231
// MI455X (gfx1250) — matrix operand order repaired and checked by reading, not compiled, not run on hardware
//
#include <hip/hip_runtime.h>

typedef __attribute__((ext_vector_type(16))) _Float16 v16h;
typedef __attribute__((ext_vector_type(8)))  float    v8f;

#define C_IN 128
#define C_OUT 128
#define EDIM 64
#define WPB 8   // waves per block (256 threads, wave32)

static __device__ __forceinline__ v8f wmma_f16(v16h a, v16h b, v8f c) {
  return __builtin_amdgcn_wmma_f32_16x16x32_f16(false, a, false, b, (short)0, c, false, false);
}

// ---------------- zero workspace region ----------------
__global__ void zero_kernel(float* p, long n) {
  long id = (long)blockIdx.x * blockDim.x + threadIdx.x;
  if (id < n) p[id] = 0.0f;
}

// ---------------- out = bias (broadcast) ----------------
__global__ void init_out_kernel(float* out, const float* bias, long n) {
  long id = (long)blockIdx.x * blockDim.x + threadIdx.x;
  if (id < n) out[id] = bias[id & (C_OUT - 1)];
}

// ---------------- W_comb = W_edge @ W_lin ; b_lin = b_edge @ W_lin ----------------
__global__ void combine_weights_kernel(const float* We, const float* Wl, const float* be,
                                       float* wcomb, float* blin) {
  int tid = blockIdx.x * blockDim.x + threadIdx.x;
  if (tid < EDIM * C_OUT) {
    int i = tid >> 7, n = tid & 127;
    float s = 0.f;
    for (int k = 0; k < C_IN; ++k) s += We[i * C_IN + k] * Wl[k * C_OUT + n];
    wcomb[tid] = s;
  } else if (tid < EDIM * C_OUT + C_OUT) {
    int n = tid - EDIM * C_OUT;
    float s = 0.f;
    for (int k = 0; k < C_IN; ++k) s += be[k] * Wl[k * C_OUT + n];
    blin[n] = s;
  }
}

// ---------------- edge scatter pass 1: S[col,d]+=attr, indeg[col]++, outdeg[row]++ ----------------
__global__ void edge_scatter_kernel(const int* row, const int* col, const float* attr,
                                    float* S, float* cnt, float* outdeg, long total) {
  long id = (long)blockIdx.x * blockDim.x + threadIdx.x;
  if (id >= total) return;
  int e = (int)(id >> 6);
  int d = (int)(id & 63);
  int c = col[e];
  atomicAdd(&S[(long)c * EDIM + d], attr[id]);
  if (d == 0) {
    atomicAdd(&cnt[c], 1.0f);
    atomicAdd(&outdeg[row[e]], 1.0f);
  }
}

// ---------------- per-node pass: loop_attr (in place over S), dinv_a, dinv_row ----------------
__global__ void node_pass_kernel(float* S_loop, const float* cnt, float* outdeg_dinv,
                                 float* dinv_a, long total) {
  long id = (long)blockIdx.x * blockDim.x + threadIdx.x;
  if (id >= total) return;
  int i = (int)(id >> 6);
  int d = (int)(id & 63);
  float s  = S_loop[id];
  float la = s / fmaxf(cnt[i], 1.0f);   // mean of incoming edge_attr
  S_loop[id] = la;                       // becomes loop_attr
  float dega = s + la;                   // segsum over col2 (edges + self loop)
  dinv_a[id] = (dega > 0.f) ? rsqrtf(dega) : 0.f;
  if (d == 0) outdeg_dinv[i] = rsqrtf(outdeg_dinv[i] + 1.0f);  // deg over row2
}

// ---------------- xlin = x @ W_lin + b_lin  (WMMA f16, f32 accum) ----------------
__global__ __launch_bounds__(256) void xlin_gemm_kernel(const float* __restrict__ x,
                                                        const float* __restrict__ Wl,
                                                        const float* __restrict__ blin,
                                                        float* __restrict__ xlin, int ntiles) {
  __shared__ v16h bfrag[4][8][32];   // [kc][ntile][lane] : W_lin as f16 B fragments, 32KB
  int tid = threadIdx.x, lane = tid & 31, wave = tid >> 5;
  // cooperative stage of W_lin (128x128) into B-fragment layout
  for (int p = wave; p < 32; p += WPB) {
    int kc = p >> 3, nt = p & 7;
    int n = nt * 16 + (lane & 15);
    int kbase = kc * 32 + ((lane < 16) ? 0 : 16);
    v16h b;
#pragma unroll
    for (int j = 0; j < 16; ++j) b[j] = (_Float16)Wl[(kbase + j) * C_OUT + n];
    bfrag[kc][nt][lane] = b;
  }
  __syncthreads();

  int tile = blockIdx.x * WPB + wave;
  if (tile >= ntiles) return;            // wave-uniform: WMMA runs with full EXEC
  int m = lane & 15, khalf = lane >> 4;
  const float* xp = x + ((long)tile * 16 + m) * (long)C_IN;

  v16h A[4];
#pragma unroll
  for (int kc = 0; kc < 4; ++kc) {
    v16h a;
#pragma unroll
    for (int j = 0; j < 16; ++j) {
      int K = kc * 32 + khalf * 16 + j;
      a[j] = (_Float16)xp[K];
    }
    A[kc] = a;
  }
#pragma unroll
  for (int nt = 0; nt < 8; ++nt) {
    v8f acc = {0.f, 0.f, 0.f, 0.f, 0.f, 0.f, 0.f, 0.f};
#pragma unroll
    for (int kc = 0; kc < 4; ++kc) acc = wmma_f16(A[kc], bfrag[kc][nt][lane], acc);
    int n = nt * 16 + (lane & 15);
    float bl = blin[n];
#pragma unroll
    for (int rr = 0; rr < 8; ++rr) {
      long rown = (long)tile * 16 + rr + khalf * 8;
      xlin[rown * C_OUT + n] = acc[rr] + bl;
    }
  }
}

// ---------------- edge message pass: per-edge K=64 WMMA GEMM + scatter-add ----------------
__global__ __launch_bounds__(256) void edge_msg_kernel(const int* __restrict__ row,
                                                       const int* __restrict__ col,
                                                       const float* __restrict__ eattr,
                                                       const float* __restrict__ loop_attr,
                                                       const float* __restrict__ dinv_a,
                                                       const float* __restrict__ dinv_row,
                                                       const float* __restrict__ xlin,
                                                       const float* __restrict__ wcomb,
                                                       float* __restrict__ out,
                                                       int E_, int N_, int ntiles) {
  __shared__ v16h bfrag[2][8][32];   // W_comb (64x128) as f16 B fragments, 16KB
  int tid = threadIdx.x, lane = tid & 31, wave = tid >> 5;
  for (int p = wave; p < 16; p += WPB) {
    int kc = p >> 3, nt = p & 7;
    int n = nt * 16 + (lane & 15);
    int kbase = kc * 32 + ((lane < 16) ? 0 : 16);
    v16h b;
#pragma unroll
    for (int j = 0; j < 16; ++j) b[j] = (_Float16)wcomb[(kbase + j) * C_OUT + n];
    bfrag[kc][nt][lane] = b;
  }
  __syncthreads();

  int tile = blockIdx.x * WPB + wave;
  if (tile >= ntiles) return;            // wave-uniform
  long E2 = (long)E_ + N_;
  int m = lane & 15, khalf = lane >> 4;

  long e = (long)tile * 16 + m;
  if (e >= E2) e = E2 - 1;               // clamp (tail tile); atomics guarded below
  int r, c;
  const float* ap;
  if (e < E_) {                          // real edge
    r = row[e]; c = col[e];
    ap = eattr + e * (long)EDIM;
  } else {                               // appended self loop: ea = loop_attr[i]
    int i = (int)(e - E_);
    r = i; c = i;
    ap = loop_attr + (long)i * EDIM;
  }
  float nm_own = dinv_row[r] * dinv_row[c];
  const float* dr = dinv_a + (long)r * EDIM;
  const float* dc = dinv_a + (long)c * EDIM;

  // A fragments: ea_norm = dinv_a[row] * ea * dinv_a[col], converted to f16
  v16h A[2];
#pragma unroll
  for (int kc = 0; kc < 2; ++kc) {
    v16h a;
#pragma unroll
    for (int j = 0; j < 16; ++j) {
      int K = kc * 32 + khalf * 16 + j;
      a[j] = (_Float16)(ap[K] * dr[K] * dc[K]);
    }
    A[kc] = a;
  }

  // exchange per-edge row/col/norm (owned by lane == edge index) via ds_bpermute
  int er8[8]; int ec8[8]; float nm8[8];
#pragma unroll
  for (int rr = 0; rr < 8; ++rr) {
    int mm = rr + khalf * 8;
    er8[rr] = __shfl(r, mm, 32);
    ec8[rr] = __shfl(c, mm, 32);
    nm8[rr] = __shfl(nm_own, mm, 32);
  }

#pragma unroll
  for (int nt = 0; nt < 8; ++nt) {
    v8f acc = {0.f, 0.f, 0.f, 0.f, 0.f, 0.f, 0.f, 0.f};
    acc = wmma_f16(A[0], bfrag[0][nt][lane], acc);
    acc = wmma_f16(A[1], bfrag[1][nt][lane], acc);
    int n = nt * 16 + (lane & 15);
#pragma unroll
    for (int rr = 0; rr < 8; ++rr) {
      int mm = rr + khalf * 8;
      long ge = (long)tile * 16 + mm;
      if (ge < E2) {
        float val = nm8[rr] * (xlin[(long)er8[rr] * C_OUT + n] + acc[rr]);
        atomicAdd(&out[(long)ec8[rr] * C_OUT + n], val);
      }
    }
  }
}

static inline int cdiv(long a, long b) { return (int)((a + b - 1) / b); }

extern "C" void kernel_launch(void* const* d_in, const int* in_sizes, int n_in,
                              void* d_out, int out_size, void* d_ws, size_t ws_size,
                              hipStream_t stream) {
  const float* x     = (const float*)d_in[0];
  const int*   ei    = (const int*)  d_in[1];
  const float* eattr = (const float*)d_in[2];
  const float* We    = (const float*)d_in[3];
  const float* be    = (const float*)d_in[4];
  const float* Wl    = (const float*)d_in[5];
  const float* bias  = (const float*)d_in[6];
  float* out = (float*)d_out;

  int N = in_sizes[0] / C_IN;
  int E = in_sizes[1] / 2;
  const int* row = ei;
  const int* col = ei + E;

  // workspace layout (floats)
  float* ws     = (float*)d_ws;
  float* S      = ws;                        // N*64  (segsum of edge_attr -> loop_attr)
  float* cnt    = S + (long)N * EDIM;        // N     (in-degree)
  float* outdeg = cnt + N;                   // N     (out-degree -> dinv_row)
  float* dinva  = outdeg + N;                // N*64
  float* xlin   = dinva + (long)N * EDIM;    // N*128
  float* wcomb  = xlin + (long)N * C_OUT;    // 64*128
  float* blin   = wcomb + EDIM * C_OUT;      // 128

  long nzero = (long)N * (EDIM + 2);
  zero_kernel<<<cdiv(nzero, 256), 256, 0, stream>>>(S, nzero);
  init_out_kernel<<<cdiv((long)N * C_OUT, 256), 256, 0, stream>>>(out, bias, (long)N * C_OUT);
  combine_weights_kernel<<<cdiv(EDIM * C_OUT + C_OUT, 256), 256, 0, stream>>>(We, Wl, be, wcomb, blin);

  long escatter = (long)E * EDIM;
  edge_scatter_kernel<<<cdiv(escatter, 256), 256, 0, stream>>>(row, col, eattr, S, cnt, outdeg, escatter);
  node_pass_kernel<<<cdiv((long)N * EDIM, 256), 256, 0, stream>>>(S, cnt, outdeg, dinva, (long)N * EDIM);

  int ntiles_x = N / 16;                               // N = 100000 -> 6250 tiles
  xlin_gemm_kernel<<<cdiv(ntiles_x, WPB), 256, 0, stream>>>(x, Wl, blin, xlin, ntiles_x);

  long E2 = (long)E + N;
  int ntiles_e = (int)((E2 + 15) / 16);                // 1.7M -> 106250 tiles
  edge_msg_kernel<<<cdiv(ntiles_e, WPB), 256, 0, stream>>>(row, col, eattr, S /*loop_attr*/,
                                                           dinva, outdeg /*dinv_row*/,
                                                           xlin, wcomb, out, E, N, ntiles_e);
}
